// Decoder_44770739094202
// MI455X (gfx1250) — matrix operand order repaired and checked by reading, not compiled, not run on hardware
//
#include <hip/hip_runtime.h>

// ---------------- types ----------------
typedef _Float16 v16h __attribute__((ext_vector_type(16)));
typedef _Float16 v8h  __attribute__((ext_vector_type(8)));
typedef float    v8f  __attribute__((ext_vector_type(8)));

// 16-byte payload type matching __builtin_amdgcn_global_load_async_to_lds_b128
typedef int b128_t __attribute__((vector_size(4 * sizeof(int))));
typedef __attribute__((address_space(1))) b128_t as1_b128;   // global
typedef __attribute__((address_space(3))) b128_t as3_b128;   // LDS

static constexpr int B_  = 16384;
static constexpr int S_  = 50;
static constexpr int D_  = 512;
static constexpr int Q_  = 2048;   // 4*D gate-packed width
static constexpr size_t HS = (size_t)B_ * D_;  // elements in one h/c buffer

#if __has_builtin(__builtin_amdgcn_global_load_async_to_lds_b128)
#define USE_ASYNC_LDS 1
#else
#define USE_ASYNC_LDS 0
#endif

__device__ __forceinline__ void wait_async0() {
#if __has_builtin(__builtin_amdgcn_s_wait_asynccnt)
    __builtin_amdgcn_s_wait_asynccnt(0);
#else
    asm volatile("s_wait_asynccnt 0x0" ::: "memory");
#endif
}

__device__ __forceinline__ float fast_tanh(float x) {
#if __has_builtin(__builtin_amdgcn_tanhf)
    return __builtin_amdgcn_tanhf(x);
#else
    return tanhf(x);
#endif
}

// ---------------- prep: transpose 512x2048 f32 -> 2048x512 f16 ----------------
__global__ __launch_bounds__(256)
void transpose_f16_kernel(const float* __restrict__ src, _Float16* __restrict__ dst) {
    int i = blockIdx.x * 256 + threadIdx.x;     // over 2048*512
    int n = i >> 9;                             // output row  (0..2047) = src column
    int k = i & 511;                            // output col  (0..511)  = src row
    dst[i] = (_Float16)src[(size_t)k * Q_ + n];
}

// ---------------- prep: state init ----------------
__global__ __launch_bounds__(256)
void init_state_kernel(const float* __restrict__ sh, const float* __restrict__ sc,
                       _Float16* __restrict__ enc16,
                       _Float16* __restrict__ hm0, _Float16* __restrict__ hy0,
                       float* __restrict__ cm, float* __restrict__ cy) {
    size_t i = (size_t)blockIdx.x * 256 + threadIdx.x;  // over B*D
    float h = sh[i];
    _Float16 hh = (_Float16)h;
    enc16[i] = hh; hm0[i] = hh; hy0[i] = hh;
    float c = sc[i];
    cm[i] = c; cy[i] = c;
}

// ---------------- prep: per-step head bias (bh + tril(ts) @ Wh[512:562]) ----------------
__global__ __launch_bounds__(256)
void tsbias_kernel(const float* __restrict__ Whm, const float* __restrict__ bhm,
                   const float* __restrict__ Why, const float* __restrict__ bhy,
                   float* __restrict__ tsbm, float* __restrict__ tsby) {
    int i = blockIdx.x * 256 + threadIdx.x;
    if (i < S_ * 30) {
        int t = i / 30, j = i % 30;
        float s = bhm[j];
        for (int q = 0; q <= t; ++q) s += Whm[(512 + q) * 30 + j];
        tsbm[i] = s;
    } else if (i < S_ * 30 + S_ * 15) {
        int k = i - S_ * 30;
        int t = k / 15, j = k % 15;
        float s = bhy[j];
        for (int q = 0; q <= t; ++q) s += Why[(512 + q) * 15 + j];
        tsby[k] = s;
    }
}

// ---------------- fused LSTM step ----------------
// z = [enc16|h16] @ [WT;UT] + b + cond@Wc (rank-3), then gates -> h/c update.
// Block: 256 threads (8 waves), tile 128 batch-rows x 64 d-cols x all 4 gates.
// Double-buffered LDS; tiles fetched with GLOBAL_LOAD_ASYNC_TO_LDS_B128 when available.
__global__ __launch_bounds__(256)
void lstm_step_kernel(const _Float16* __restrict__ enc16,   // [B][512]
                      const _Float16* __restrict__ hin16,   // [B][512]
                      _Float16* __restrict__ hout16,        // [B][512]
                      float* __restrict__ cstate,           // [B][512]
                      const _Float16* __restrict__ WT16,    // [2048][512]  (W[0:512]^T)
                      const _Float16* __restrict__ UT16,    // [2048][512]  (U^T)
                      const float* __restrict__ bias,       // [2048]
                      const float* __restrict__ Wfull,      // [515][2048] rows 512..514 = cond rows
                      const float* __restrict__ cond,       // [B][50][3]
                      int t) {
    __shared__ _Float16 As[2][128][40];      // ping-pong: 128 rows x 32 k (+8 pad)
    __shared__ _Float16 Bs[2][4][64][40];    // ping-pong: per gate 64 n x 32 k (+8 pad)

    const int tid  = threadIdx.x;
    const int lane = tid & 31;
    const int wave = tid >> 5;               // 0..7
    const int wm   = wave >> 1;              // 0..3 -> 32-row slab
    const int wn   = wave & 1;               // 0..1 -> 32-col slab
    const int bM   = blockIdx.x * 128;       // batch-row base
    const int bN   = blockIdx.y * 64;        // d-col base (within 512)

    v8f acc[16];                             // [gate][ms][ns] flattened
#pragma unroll
    for (int i = 0; i < 16; ++i)
#pragma unroll
        for (int e = 0; e < 8; ++e) acc[i][e] = 0.0f;

    auto issue_tile = [&](int kk, int p) {
        const bool first = kk < 16;
        const int k0 = (first ? kk : kk - 16) * 32;
        const _Float16* __restrict__ Asrc = first ? enc16 : hin16;
        const _Float16* __restrict__ Bsrc = first ? WT16  : UT16;
        // A tile: 128 rows x 32 f16 = 512 chunks of 8 f16 (16B)
#pragma unroll
        for (int u = tid; u < 512; u += 256) {
            int row = u >> 2, seg = u & 3;
            const _Float16* gp = &Asrc[(size_t)(bM + row) * D_ + k0 + seg * 8];
            _Float16* lp = &As[p][row][seg * 8];
#if USE_ASYNC_LDS
            __builtin_amdgcn_global_load_async_to_lds_b128(
                (as1_b128*)gp, (as3_b128*)lp, 0, 0);
#else
            *(v8h*)lp = *(const v8h*)gp;
#endif
        }
        // B tiles: 4 gates x 64 n x 32 f16 = 1024 chunks of 8 f16
#pragma unroll
        for (int u = tid; u < 1024; u += 256) {
            int g = u >> 8, n = (u >> 2) & 63, seg = u & 3;
            const _Float16* gp = &Bsrc[(size_t)(g * D_ + bN + n) * D_ + k0 + seg * 8];
            _Float16* lp = &Bs[p][g][n][seg * 8];
#if USE_ASYNC_LDS
            __builtin_amdgcn_global_load_async_to_lds_b128(
                (as1_b128*)gp, (as3_b128*)lp, 0, 0);
#else
            *(v8h*)lp = *(const v8h*)gp;
#endif
        }
    };

    const int c0A = (lane < 16) ? 0 : 8;   // A frag: K chunks {c0A..+7, c0A+16..+23}
    const int c0B = (lane < 16) ? 0 : 8;   // B frag: K chunks {c0B..+7, c0B+16..+23}
    const int l15 = lane & 15;

    issue_tile(0, 0);

    for (int kk = 0; kk < 32; ++kk) {      // K = 1024 = [enc|h] x 512, 32 per stage
        const int p = kk & 1;
#if USE_ASYNC_LDS
        wait_async0();                     // my share of tile kk is in LDS
#endif
        __syncthreads();                   // everyone's share is in LDS; prev readers done
        if (kk < 31) issue_tile(kk + 1, p ^ 1);   // DMA next tile under the WMMAs

        v16h afr[2];
#pragma unroll
        for (int ms = 0; ms < 2; ++ms) {
            int m = wm * 32 + ms * 16 + l15;
            v8h lo = *(const v8h*)&As[p][m][c0A];
            v8h hi = *(const v8h*)&As[p][m][c0A + 16];
            afr[ms] = __builtin_shufflevector(lo, hi,
                0,1,2,3,4,5,6,7,8,9,10,11,12,13,14,15);
        }
        v16h bfr[4][2];
#pragma unroll
        for (int g = 0; g < 4; ++g)
#pragma unroll
            for (int ns = 0; ns < 2; ++ns) {
                int n = wn * 32 + ns * 16 + l15;
                v8h lo = *(const v8h*)&Bs[p][g][n][c0B];
                v8h hi = *(const v8h*)&Bs[p][g][n][c0B + 16];
                bfr[g][ns] = __builtin_shufflevector(lo, hi,
                    0,1,2,3,4,5,6,7,8,9,10,11,12,13,14,15);
            }
#pragma unroll
        for (int g = 0; g < 4; ++g)
#pragma unroll
            for (int ms = 0; ms < 2; ++ms)
#pragma unroll
                for (int ns = 0; ns < 2; ++ns) {
                    int i = (g * 2 + ms) * 2 + ns;
                    acc[i] = __builtin_amdgcn_wmma_f32_16x16x32_f16(
                        false, afr[ms], false, bfr[g][ns],
                        (short)0, acc[i], false, false);
                }
    }

    // ---- fused LSTM epilogue: + bias + rank-3 cond correction, gates, h/c ----
    const int rh = (lane >> 4) & 1;        // C/D layout: lanes 16-31 hold M = r+8
#pragma unroll
    for (int ms = 0; ms < 2; ++ms)
#pragma unroll
        for (int r = 0; r < 8; ++r) {
            int b = bM + wm * 32 + ms * 16 + rh * 8 + r;
            const float* cb = &cond[((size_t)b * S_ + t) * 3];
            float cx = cb[0], cy = cb[1], cz = cb[2];
#pragma unroll
            for (int ns = 0; ns < 2; ++ns) {
                int d = bN + wn * 32 + ns * 16 + l15;
                float z[4];
#pragma unroll
                for (int g = 0; g < 4; ++g) {
                    int col = g * D_ + d;
                    z[g] = acc[(g * 2 + ms) * 2 + ns][r] + bias[col]
                         + cx * Wfull[512 * Q_ + col]
                         + cy * Wfull[513 * Q_ + col]
                         + cz * Wfull[514 * Q_ + col];
                }
                float si = 1.0f / (1.0f + __expf(-z[0]));
                float sf = 1.0f / (1.0f + __expf(-z[1]));
                float so = 1.0f / (1.0f + __expf(-z[3]));
                size_t idx = (size_t)b * D_ + d;
                float cnew = sf * cstate[idx] + si * fast_tanh(z[2]);
                float hnew = so * fast_tanh(cnew);
                cstate[idx] = cnew;
                hout16[idx] = (_Float16)hnew;
            }
        }
}

// ---------------- MDN heads: wave-per-batch-row, shfl_xor reduce ----------------
__global__ __launch_bounds__(256)
void head_kernel(const _Float16* __restrict__ hm16, const _Float16* __restrict__ hy16,
                 const float* __restrict__ Whm, const float* __restrict__ Why,
                 const float* __restrict__ tsbm, const float* __restrict__ tsby,
                 float* __restrict__ outm, float* __restrict__ outy, int t) {
    int b    = (blockIdx.x * 256 + threadIdx.x) >> 5;   // global wave id = batch row
    int lane = threadIdx.x & 31;

    float am[30], ay[15];
#pragma unroll
    for (int j = 0; j < 30; ++j) am[j] = 0.0f;
#pragma unroll
    for (int j = 0; j < 15; ++j) ay[j] = 0.0f;

    for (int d = lane; d < D_; d += 32) {
        float hv = (float)hm16[(size_t)b * D_ + d];
        const float* wr = &Whm[d * 30];
#pragma unroll
        for (int j = 0; j < 30; ++j) am[j] += hv * wr[j];
        float yv = (float)hy16[(size_t)b * D_ + d];
        const float* wy = &Why[d * 15];
#pragma unroll
        for (int j = 0; j < 15; ++j) ay[j] += yv * wy[j];
    }
#pragma unroll
    for (int off = 16; off > 0; off >>= 1) {
#pragma unroll
        for (int j = 0; j < 30; ++j) am[j] += __shfl_xor(am[j], off, 32);
#pragma unroll
        for (int j = 0; j < 15; ++j) ay[j] += __shfl_xor(ay[j], off, 32);
    }
    if (lane == 0) {
#pragma unroll
        for (int j = 0; j < 30; ++j) am[j] += tsbm[t * 30 + j];
#pragma unroll
        for (int j = 0; j < 15; ++j) ay[j] += tsby[t * 15 + j];

        float* om = outm + ((size_t)b * S_ + t) * 30;
        float mx = am[0];
#pragma unroll
        for (int j = 1; j < 5; ++j) mx = fmaxf(mx, am[j]);
        float es[5], s = 0.0f;
#pragma unroll
        for (int j = 0; j < 5; ++j) { es[j] = __expf(am[j] - mx); s += es[j]; }
        float inv = 1.0f / s;
#pragma unroll
        for (int j = 0; j < 5; ++j) om[j]      = es[j] * inv;
#pragma unroll
        for (int j = 0; j < 5; ++j) om[5 + j]  = am[5 + j];
#pragma unroll
        for (int j = 0; j < 5; ++j) om[10 + j] = __expf(am[10 + j]);
#pragma unroll
        for (int j = 0; j < 5; ++j) om[15 + j] = am[15 + j];
#pragma unroll
        for (int j = 0; j < 5; ++j) om[20 + j] = __expf(am[20 + j]);
#pragma unroll
        for (int j = 0; j < 5; ++j) om[25 + j] = fast_tanh(am[25 + j]);

        float* oy = outy + ((size_t)b * S_ + t) * 15;
        float my = ay[0];
#pragma unroll
        for (int j = 1; j < 5; ++j) my = fmaxf(my, ay[j]);
        float ey[5], sy = 0.0f;
#pragma unroll
        for (int j = 0; j < 5; ++j) { ey[j] = __expf(ay[j] - my); sy += ey[j]; }
        float invy = 1.0f / sy;
#pragma unroll
        for (int j = 0; j < 5; ++j) oy[j]      = ey[j] * invy;
#pragma unroll
        for (int j = 0; j < 5; ++j) oy[5 + j]  = ay[5 + j];
#pragma unroll
        for (int j = 0; j < 5; ++j) oy[10 + j] = __expf(ay[10 + j]);
    }
}

// ---------------- host ----------------
extern "C" void kernel_launch(void* const* d_in, const int* in_sizes, int n_in,
                              void* d_out, int out_size, void* d_ws, size_t ws_size,
                              hipStream_t stream) {
    (void)in_sizes; (void)n_in; (void)out_size; (void)ws_size;
    const float* conditions = (const float*)d_in[0];
    const float* state_h    = (const float*)d_in[1];
    const float* state_c    = (const float*)d_in[2];
    const float* W_m  = (const float*)d_in[3];
    const float* U_m  = (const float*)d_in[4];
    const float* b_m  = (const float*)d_in[5];
    const float* Wh_m = (const float*)d_in[6];
    const float* bh_m = (const float*)d_in[7];
    const float* W_y  = (const float*)d_in[8];
    const float* U_y  = (const float*)d_in[9];
    const float* b_y  = (const float*)d_in[10];
    const float* Wh_y = (const float*)d_in[11];
    const float* bh_y = (const float*)d_in[12];

    float* outm = (float*)d_out;
    float* outy = outm + (size_t)B_ * S_ * 30;

    // workspace carve-up (256B aligned)
    char* w = (char*)d_ws;
    auto alloc = [&](size_t bytes) {
        void* p = (void*)w;
        w += (bytes + 255) & ~(size_t)255;
        return p;
    };
    _Float16* UT_m  = (_Float16*)alloc((size_t)Q_ * D_ * 2);
    _Float16* UT_y  = (_Float16*)alloc((size_t)Q_ * D_ * 2);
    _Float16* WT_m  = (_Float16*)alloc((size_t)Q_ * D_ * 2);
    _Float16* WT_y  = (_Float16*)alloc((size_t)Q_ * D_ * 2);
    _Float16* enc16 = (_Float16*)alloc(HS * 2);
    _Float16* hm16  = (_Float16*)alloc(2 * HS * 2);   // ping-pong
    _Float16* hy16  = (_Float16*)alloc(2 * HS * 2);   // ping-pong
    float*    cm    = (float*)alloc(HS * 4);
    float*    cy    = (float*)alloc(HS * 4);
    float*    tsbm  = (float*)alloc((size_t)S_ * 30 * 4);
    float*    tsby  = (float*)alloc((size_t)S_ * 15 * 4);

    const int TGRID = (Q_ * D_) / 256;   // 4096
    transpose_f16_kernel<<<TGRID, 256, 0, stream>>>(U_m, UT_m);
    transpose_f16_kernel<<<TGRID, 256, 0, stream>>>(U_y, UT_y);
    transpose_f16_kernel<<<TGRID, 256, 0, stream>>>(W_m, WT_m);  // rows 0..511 of W
    transpose_f16_kernel<<<TGRID, 256, 0, stream>>>(W_y, WT_y);
    init_state_kernel<<<(int)(HS / 256), 256, 0, stream>>>(
        state_h, state_c, enc16, hm16, hy16, cm, cy);
    tsbias_kernel<<<9, 256, 0, stream>>>(Wh_m, bh_m, Wh_y, bh_y, tsbm, tsby);

    dim3 grid(B_ / 128, D_ / 64);   // 128 x 8
    for (int t = 0; t < S_; ++t) {
        const _Float16* hin_m  = hm16 + (size_t)(t & 1) * HS;
        _Float16*       hout_m = hm16 + (size_t)((t + 1) & 1) * HS;
        const _Float16* hin_y  = hy16 + (size_t)(t & 1) * HS;
        _Float16*       hout_y = hy16 + (size_t)((t + 1) & 1) * HS;
        lstm_step_kernel<<<grid, 256, 0, stream>>>(
            enc16, hin_m, hout_m, cm, WT_m, UT_m, b_m, W_m, conditions, t);
        lstm_step_kernel<<<grid, 256, 0, stream>>>(
            enc16, hin_y, hout_y, cy, WT_y, UT_y, b_y, W_y, conditions, t);
        head_kernel<<<B_ / 8, 256, 0, stream>>>(
            hout_m, hout_y, Wh_m, Wh_y, tsbm, tsby, outm, outy, t);
    }
}
